// SegmentationPointNetP2_16844861734922
// MI455X (gfx1250) — matrix operand order repaired and checked by reading, not compiled, not run on hardware
//
#include <hip/hip_runtime.h>
#include <hip/hip_bf16.h>

// ---------------------------------------------------------------------------
// PointNet++ segmentation pipeline for gfx1250 (MI455X).
// GEMMs use v_wmma_f32_16x16x32_f16 (wave32, f32 accumulate), 16x64 output
// strip per wave, fully-unrolled K loop (compile-time KSTEPS) so the compiler
// register-renames the double buffer instead of emitting mov chains.
// ---------------------------------------------------------------------------

typedef _Float16 h16;
typedef _Float16 v16h __attribute__((ext_vector_type(16)));
typedef _Float16 v8h  __attribute__((ext_vector_type(8)));
typedef float    v8f  __attribute__((ext_vector_type(8)));

// ======================= Farthest Point Sampling ===========================
// One block per call; serial chain of nS argmax steps. min-distances live in
// registers (8 points per thread, P <= 8192), argmax via LDS tree reduction.
__global__ void __launch_bounds__(1024)
fps_kernel(const float* __restrict__ pos, int P, int nS,
           int* __restrict__ outIdx, float* __restrict__ outPos)
{
  const int tid = threadIdx.x;
  __shared__ float sV[1024];
  __shared__ int   sI[1024];
  __shared__ float sLP[3];
  __shared__ int   sCur;
  float mind[8];
#pragma unroll
  for (int j = 0; j < 8; ++j) mind[j] = 3.4e38f;
  if (tid == 0) sCur = 0;
  __syncthreads();
  for (int s = 0; s < nS; ++s) {
    if (tid == 0) {
      int c = sCur;
      outIdx[s] = c;
      float px = pos[c*3+0], py = pos[c*3+1], pz = pos[c*3+2];
      sLP[0] = px; sLP[1] = py; sLP[2] = pz;
      outPos[s*3+0] = px; outPos[s*3+1] = py; outPos[s*3+2] = pz;
    }
    __syncthreads();
    float lx = sLP[0], ly = sLP[1], lz = sLP[2];
    float best = -1.0f; int bi = 0x7fffffff;
#pragma unroll
    for (int j = 0; j < 8; ++j) {
      int p = tid + j*1024;
      if (p < P) {
        float dx = pos[p*3+0]-lx, dy = pos[p*3+1]-ly, dz = pos[p*3+2]-lz;
        float d = dx*dx + dy*dy + dz*dz;
        if (d < mind[j]) mind[j] = d;
        if (mind[j] > best) { best = mind[j]; bi = p; }
      }
    }
    sV[tid] = best; sI[tid] = bi;
    __syncthreads();
    for (int offr = 512; offr > 0; offr >>= 1) {
      if (tid < offr) {
        float v2 = sV[tid+offr]; int i2 = sI[tid+offr];
        if (v2 > sV[tid] || (v2 == sV[tid] && i2 < sI[tid])) { sV[tid] = v2; sI[tid] = i2; }
      }
      __syncthreads();
    }
    if (tid == 0) sCur = sI[0];
    __syncthreads();
  }
}

// ======================= brute-force 16-NN (ball query) ====================
// One thread per sampled point; register-resident sorted top-16 (swap chain,
// all indices compile-time so arrays stay in VGPRs). Source points are
// L2-resident (96 KB per graph << 192 MB L2).
__global__ void knn16_kernel(const float* __restrict__ posS, int nS,
                             const float* __restrict__ pos, int P,
                             int* __restrict__ nbr, float* __restrict__ nd2)
{
  int s = blockIdx.x*blockDim.x + threadIdx.x;
  if (s >= nS) return;
  float sx = posS[s*3+0], sy = posS[s*3+1], sz = posS[s*3+2];
  float bd[16]; int bi[16];
#pragma unroll
  for (int k = 0; k < 16; ++k) { bd[k] = 3.4e38f; bi[k] = 0; }
  for (int p = 0; p < P; ++p) {
    float dx = pos[p*3+0]-sx, dy = pos[p*3+1]-sy, dz = pos[p*3+2]-sz;
    float d = dx*dx + dy*dy + dz*dz;
    if (d < bd[15]) {
      float cd = d; int ci = p;
#pragma unroll
      for (int k = 0; k < 16; ++k) {
        if (cd < bd[k]) { float td = bd[k]; int ti = bi[k];
                          bd[k] = cd; bi[k] = ci; cd = td; ci = ti; }
      }
    }
  }
#pragma unroll
  for (int k = 0; k < 16; ++k) { nbr[s*16+k] = bi[k]; nd2[s*16+k] = bd[k]; }
}

// ======================= 3-NN for knn_interpolate ==========================
__global__ void knn3_kernel(const float* __restrict__ posD, int nD,
                            const float* __restrict__ posS, int nS,
                            int* __restrict__ idx3, float* __restrict__ w3)
{
  int d = blockIdx.x*blockDim.x + threadIdx.x;
  if (d >= nD) return;
  float dx0 = posD[d*3+0], dy0 = posD[d*3+1], dz0 = posD[d*3+2];
  float bd0 = 3.4e38f, bd1 = 3.4e38f, bd2 = 3.4e38f;
  int   b0 = 0, b1 = 0, b2 = 0;
  for (int p = 0; p < nS; ++p) {
    float ex = posS[p*3+0]-dx0, ey = posS[p*3+1]-dy0, ez = posS[p*3+2]-dz0;
    float dd = ex*ex + ey*ey + ez*ez;
    if (dd < bd2) {
      if (dd < bd0)      { bd2 = bd1; b2 = b1; bd1 = bd0; b1 = b0; bd0 = dd; b0 = p; }
      else if (dd < bd1) { bd2 = bd1; b2 = b1; bd1 = dd;  b1 = p; }
      else               { bd2 = dd;  b2 = p; }
    }
  }
  float w0 = 1.0f/fmaxf(bd0, 1e-16f);
  float w1 = 1.0f/fmaxf(bd1, 1e-16f);
  float w2 = 1.0f/fmaxf(bd2, 1e-16f);
  float ws = w0 + w1 + w2;
  idx3[d*3+0] = b0; idx3[d*3+1] = b1; idx3[d*3+2] = b2;
  w3[d*3+0] = w0/ws; w3[d*3+1] = w1/ws; w3[d*3+2] = w2/ws;
}

// ======================= misc elementwise kernels ==========================
__global__ void zero16_kernel(h16* __restrict__ p, int n)
{
  int t = blockIdx.x*blockDim.x + threadIdx.x;
  if (t < n) p[t] = (h16)0.0f;
}

// copy f16 [M,C] (leading dim srcld; srcld==0 broadcasts row 0) into dst cols [c0,c0+C)
__global__ void copy16_kernel(const h16* __restrict__ src, int srcld,
                              h16* __restrict__ dst, int ldd, int c0, int M, int C)
{
  int t = blockIdx.x*blockDim.x + threadIdx.x;
  if (t >= M*C) return;
  int r = t / C, c = t - r*C;
  dst[(size_t)r*ldd + c0 + c] = src[(size_t)r*srcld + c];
}

__global__ void copy32to16_kernel(const float* __restrict__ src, int srcld,
                                  h16* __restrict__ dst, int ldd, int c0, int M, int C)
{
  int t = blockIdx.x*blockDim.x + threadIdx.x;
  if (t >= M*C) return;
  int r = t / C, c = t - r*C;
  dst[(size_t)r*ldd + c0 + c] = (h16)src[(size_t)r*srcld + c];
}

// Build SA input rows [nS*K, Cf+3]: cat(x[nbr], pos[nbr]-pos_s). feat is f32 or f16.
__global__ void gather_sa_kernel(const float* __restrict__ feat32,
                                 const h16* __restrict__ feat16, int Cf,
                                 const float* __restrict__ posSrc,
                                 const float* __restrict__ posS,
                                 const int* __restrict__ nbr, int nS, int Kn,
                                 h16* __restrict__ out, int ldo)
{
  int cw = Cf + 3;
  int total = nS * Kn * cw;
  int t = blockIdx.x*blockDim.x + threadIdx.x;
  if (t >= total) return;
  int row = t / cw, c = t - row*cw;
  int s = row / Kn;
  int j = nbr[row];
  float v;
  if (c < Cf) v = feat32 ? feat32[(size_t)j*Cf + c] : (float)feat16[(size_t)j*Cf + c];
  else        v = posSrc[j*3 + (c - Cf)] - posS[s*3 + (c - Cf)];
  out[(size_t)row*ldo + c] = (h16)v;
}

// Radius-masked max aggregation over K neighbors (LN already applied, f32 in).
__global__ void maxagg_kernel(const float* __restrict__ h, const float* __restrict__ d2,
                              int nS, int Kn, int C, float R2, h16* __restrict__ out)
{
  int t = blockIdx.x*blockDim.x + threadIdx.x;
  if (t >= nS*C) return;
  int s = t / C, c = t - s*C;
  float m = -1e30f;
  for (int k = 0; k < Kn; ++k)
    if (d2[s*Kn+k] < R2) m = fmaxf(m, h[(size_t)(s*Kn+k)*C + c]);
  out[t] = (h16)m;
}

// Global max pool over M rows.
__global__ void gmax_kernel(const float* __restrict__ h, int M, int C, h16* __restrict__ out)
{
  int c = blockIdx.x*blockDim.x + threadIdx.x;
  if (c >= C) return;
  float m = -1e30f;
  for (int r = 0; r < M; ++r) m = fmaxf(m, h[(size_t)r*C + c]);
  out[c] = (h16)m;
}

// knn_interpolate apply: dst[r, c0+c] = sum_j w[r,j] * src[idx[r,j], c]
__global__ void interp3_kernel(const h16* __restrict__ src, int C,
                               const int* __restrict__ idx3, const float* __restrict__ w3,
                               int nD, h16* __restrict__ dst, int ldd, int c0)
{
  int t = blockIdx.x*blockDim.x + threadIdx.x;
  if (t >= nD*C) return;
  int r = t / C, c = t - r*C;
  float acc = 0.0f;
#pragma unroll
  for (int j = 0; j < 3; ++j)
    acc += w3[r*3+j] * (float)src[(size_t)idx3[r*3+j]*C + c];
  dst[(size_t)r*ldd + c0 + c] = (h16)acc;
}

// LayerNorm over feature dim; one wave per row (wave32). Optional f32/f16 out.
__global__ void __launch_bounds__(256)
ln_kernel(const float* __restrict__ in, int M, int C,
          const float* __restrict__ g, const float* __restrict__ bt,
          float* __restrict__ out32, h16* __restrict__ out16)
{
  int row  = blockIdx.x * 8 + (threadIdx.x >> 5);
  int lane = threadIdx.x & 31;
  if (row >= M) return;
  const float* r = in + (size_t)row*C;
  float s = 0.0f, s2 = 0.0f;
  for (int c = lane; c < C; c += 32) { float v = r[c]; s += v; s2 += v*v; }
#pragma unroll
  for (int o = 16; o > 0; o >>= 1) { s += __shfl_xor(s, o, 32); s2 += __shfl_xor(s2, o, 32); }
  float mu  = s / (float)C;
  float var = fmaxf(s2 / (float)C - mu*mu, 0.0f);
  float inv = rsqrtf(var + 1e-5f);
  for (int c = lane; c < C; c += 32) {
    float v = (r[c] - mu) * inv * g[c] + bt[c];
    if (out16) out16[(size_t)row*C + c] = (h16)v;
    else       out32[(size_t)row*C + c] = v;
  }
}

// ======================= WMMA GEMM =========================================
// C[M,Npad] = act(A[M,Kd] * Wt^T + bias). A f16 row-major (lda=Kd=KS*32),
// Wt f16 [Npad,Kd] row-major (= W transposed), f32 accumulate via
// v_wmma_f32_16x16x32_f16.
//
// Each wave owns a 16 x (16*NT) output strip: one A fragment feeds NT
// back-to-back WMMAs (matrix-op burst; XDL WMMA co-executes with VALU).
// The K loop is fully unrolled (KS is a template constant), so every
// fragment lives in its own virtual register and is consumed directly by its
// WMMA -- the scheduler hoists the next step's global_load_b128s above the
// current WMMAs and emits partial s_wait_loadcnt, with no mov rotation.
//
// Fragment layouts used here (A and B share contiguous k = lg*16+e, not the ISA table order):
//   A (16x32 f16): lane l (lm=l&15,lg=l>>4) = row m+lm; halves e = K[lg*16+e]
//                  -> two 16B loads.
//   B (32x16 f16): lane l = col n+lm; halves e = K[lg*16+e] -> one 32B load of Wt row.
//   C/D (16x16 f32): acc[r] = C[m + lg*8 + r][n + lm].
template<int NT, int KS>
__global__ void __launch_bounds__(256)
gemm_kernel(const h16* __restrict__ A,
            const h16* __restrict__ Wt,
            const float* __restrict__ bias,
            int M, int Npad, int Nreal,
            h16* __restrict__ out16, float* __restrict__ out32, int ldo,
            int relu)
{
  constexpr int Kd = KS * 32;
  const int nGroups = Npad / (16*NT);
  const int mTiles  = M >> 4;
  int wid  = (blockIdx.x * blockDim.x + threadIdx.x) >> 5;
  int lane = threadIdx.x & 31;
  if (wid >= mTiles * nGroups) return;           // wave-uniform exit
  int mt = wid / nGroups, ng = wid - mt * nGroups;
  int m = mt << 4, n0 = ng * (16*NT);
  int lm = lane & 15, lg = lane >> 4;

  const h16* arow = A + (size_t)(m + lm) * Kd + lg * 16;
  const h16* brow[NT];
#pragma unroll
  for (int t = 0; t < NT; ++t)
    brow[t] = Wt + (size_t)(n0 + t*16 + lm) * Kd + lg * 16;

  const v8f vzero = {};
  v8f acc[NT];
#pragma unroll
  for (int t = 0; t < NT; ++t) acc[t] = vzero;

#pragma unroll
  for (int ks = 0; ks < KS; ++ks) {
    const int k0 = ks * 32;
    v8h alo = *(const v8h*)(arow + k0);
    v8h ahi = *(const v8h*)(arow + k0 + 8);
    v16h a;
#pragma unroll
    for (int i = 0; i < 8; ++i) { a[i] = alo[i]; a[i+8] = ahi[i]; }
#pragma unroll
    for (int t = 0; t < NT; ++t) {
      v16h bfrag = *(const v16h*)(brow[t] + k0);
      acc[t] = __builtin_amdgcn_wmma_f32_16x16x32_f16(false, a, false, bfrag,
                                                      (short)0, acc[t], false, false);
    }
  }

#pragma unroll
  for (int t = 0; t < NT; ++t) {
    int col = n0 + t*16 + lm;
    if (col < Nreal) {
      float bv = bias[col];
#pragma unroll
      for (int r = 0; r < 8; ++r) {
        int row = m + lg*8 + r;
        float v = acc[t][r] + bv;
        if (relu) v = fmaxf(v, 0.0f);
        if (out16) out16[(size_t)row*ldo + col] = (h16)v;
        else       out32[(size_t)row*ldo + col] = v;
      }
    }
  }
}

// f32 W[K,N] -> f16 Wt[Npad,Kdp] (transposed, zero-padded).
__global__ void prep_w_kernel(const float* __restrict__ W, int K, int N,
                              int Kdp, int Npad, h16* __restrict__ Wt)
{
  int t = blockIdx.x*blockDim.x + threadIdx.x;
  if (t >= Npad*Kdp) return;
  int n = t / Kdp, k = t - n*Kdp;
  float v = (n < N && k < K) ? W[(size_t)k*N + n] : 0.0f;
  Wt[t] = (h16)v;
}

// ======================= host orchestration ================================
struct WDesc { int wi, bi, K, N, Kdp, Npad; };
// params flattened in jax-pytree order (sorted dict keys): fp1,fp2,fp3,head,sa1,sa2,sa3;
// within each module: W0..Wn, b0..bn, bt, g.  d_in[0]=x, d_in[1]=pos.
static const WDesc g_wd[22] = {
  /* sa1 */ {34,37,  11,  64,  32,  64}, {35,38,  64,  64,  64,  64}, {36,39,  64, 128,  64, 128},
  /* sa2 */ {42,45, 131, 128, 160, 128}, {43,46, 128, 128, 128, 128}, {44,47, 128, 256, 128, 256},
  /* sa3 */ {50,53, 259, 512, 288, 512}, {51,54, 512, 512, 512, 512}, {52,55, 512,1024, 512,1024},
  /* fp1 */ { 2, 5,1280, 256,1280, 256}, { 3, 6, 256, 256, 256, 256}, { 4, 7, 256, 256, 256, 256},
  /* fp2 */ {10,13, 384, 256, 384, 256}, {11,14, 256, 256, 256, 256}, {12,15, 256, 128, 256, 128},
  /* fp3 */ {18,21, 136, 128, 160, 128}, {19,22, 128, 128, 128, 128}, {20,23, 128, 128, 128, 128},
  /* head*/ {26,30, 128, 128, 128, 128}, {27,31, 128, 128, 128, 128}, {28,32, 128, 128, 128, 128},
            {29,33, 128,   3, 128,  16},
};

static inline int cdiv(int a, int b) { return (a + b - 1) / b; }

extern "C" void kernel_launch(void* const* d_in, const int* in_sizes, int n_in,
                              void* d_out, int out_size, void* d_ws, size_t ws_size,
                              hipStream_t stream) {
  (void)in_sizes; (void)n_in; (void)out_size; (void)ws_size;
  const int B = 4, P = 8000, N1 = 1600, N2 = 400, Kn = 16;
  const float R2 = 1000.0f * 1000.0f;

  const float* x_all   = (const float*)d_in[0];
  const float* pos_all = (const float*)d_in[1];
  float* out = (float*)d_out;

  // ---- workspace carve-up (deterministic; ~25 MB) ----
  size_t off = 0;
  auto alloc = [&](size_t bytes) -> void* {
    void* p = (void*)((char*)d_ws + off);
    off += (bytes + 255) & ~(size_t)255;
    return p;
  };
  h16* wt[22];
  for (int i = 0; i < 22; ++i)
    wt[i] = (h16*)alloc((size_t)g_wd[i].Npad * g_wd[i].Kdp * sizeof(h16));
  int*   idx1 = (int*)  alloc((size_t)N1*4);
  float* pos1 = (float*)alloc((size_t)N1*3*4);
  int*   idx2 = (int*)  alloc((size_t)N2*4);
  float* pos2 = (float*)alloc((size_t)N2*3*4);
  int*   nbr1 = (int*)  alloc((size_t)N1*Kn*4);
  float* d21  = (float*)alloc((size_t)N1*Kn*4);
  int*   nbr2 = (int*)  alloc((size_t)N2*Kn*4);
  float* d22  = (float*)alloc((size_t)N2*Kn*4);
  int*   k3iA = (int*)  alloc((size_t)N1*3*4);
  float* k3wA = (float*)alloc((size_t)N1*3*4);
  int*   k3iB = (int*)  alloc((size_t)P*3*4);
  float* k3wB = (float*)alloc((size_t)P*3*4);
  h16*   x1   = (h16*)  alloc((size_t)N1*128*sizeof(h16));
  h16*   x2   = (h16*)  alloc((size_t)N2*256*sizeof(h16));
  h16*   x3   = (h16*)  alloc((size_t)1024*sizeof(h16));
  h16*   f1   = (h16*)  alloc((size_t)N2*256*sizeof(h16));
  h16*   f2   = (h16*)  alloc((size_t)N1*128*sizeof(h16));
  const size_t BUFE = (size_t)25600*64;           // max f16 activation elems
  h16*   bufA = (h16*)  alloc(BUFE*sizeof(h16));
  h16*   bufB = (h16*)  alloc(BUFE*sizeof(h16));
  float* C32  = (float*)alloc((size_t)25600*128*4); // max f32 pre-LN elems

  // ---- weight prep: f32 [K,N] -> f16 [Npad,Kdp] (transposed, padded) ----
  for (int i = 0; i < 22; ++i) {
    const WDesc& d = g_wd[i];
    int n = d.Npad * d.Kdp;
    prep_w_kernel<<<cdiv(n,256),256,0,stream>>>((const float*)d_in[d.wi],
                                                d.K, d.N, d.Kdp, d.Npad, wt[i]);
  }

  auto gemm = [&](const h16* A, int li, int M, h16* o16, float* o32, int ldo, int relu) {
    const WDesc& d = g_wd[li];
    const float* bias = (const float*)d_in[d.bi];
    int ks = d.Kdp / 32;
    if ((d.Npad & 63) == 0) {
      int grid = cdiv((M/16) * (d.Npad/64), 8);
#define GEMM4(KSV) gemm_kernel<4,KSV><<<grid,256,0,stream>>>(A, wt[li], bias, M, d.Npad, d.N, o16, o32, ldo, relu)
      switch (ks) {
        case 1:  GEMM4(1);  break;
        case 2:  GEMM4(2);  break;
        case 4:  GEMM4(4);  break;
        case 5:  GEMM4(5);  break;
        case 8:  GEMM4(8);  break;
        case 9:  GEMM4(9);  break;
        case 12: GEMM4(12); break;
        case 16: GEMM4(16); break;
        case 40: GEMM4(40); break;
        default: break;
      }
#undef GEMM4
    } else {
      // head final layer: Npad=16, Kd=128
      int grid = cdiv((M/16) * (d.Npad/16), 8);
      gemm_kernel<1,4><<<grid,256,0,stream>>>(A, wt[li], bias, M, d.Npad, d.N,
                                              o16, o32, ldo, relu);
    }
  };

  for (int b = 0; b < B; ++b) {
    const float* pos0 = pos_all + (size_t)b*P*3;
    const float* x0   = x_all   + (size_t)b*P*8;

    // ---- sampling + neighborhoods ----
    fps_kernel<<<1,1024,0,stream>>>(pos0, P,  N1, idx1, pos1);
    fps_kernel<<<1,1024,0,stream>>>(pos1, N1, N2, idx2, pos2);
    knn16_kernel<<<cdiv(N1,256),256,0,stream>>>(pos1, N1, pos0, P,  nbr1, d21);
    knn16_kernel<<<cdiv(N2,256),256,0,stream>>>(pos2, N2, pos1, N1, nbr2, d22);

    // ---- SA1: rows=25600, 11(->32)->64->64->128, LN, masked max -> x1 ----
    int r1 = N1*Kn;
    zero16_kernel<<<cdiv(r1*32,256),256,0,stream>>>(bufA, r1*32);
    gather_sa_kernel<<<cdiv(r1*11,256),256,0,stream>>>(x0, nullptr, 8, pos0, pos1,
                                                       nbr1, N1, Kn, bufA, 32);
    gemm(bufA, 0, r1, bufB, nullptr, 64, 1);
    gemm(bufB, 1, r1, bufA, nullptr, 64, 1);
    gemm(bufA, 2, r1, nullptr, C32, 128, 0);
    ln_kernel<<<cdiv(r1,8),256,0,stream>>>(C32, r1, 128,
        (const float*)d_in[41], (const float*)d_in[40], C32, nullptr);
    maxagg_kernel<<<cdiv(N1*128,256),256,0,stream>>>(C32, d21, N1, Kn, 128, R2, x1);

    // ---- SA2: rows=6400, 131(->160)->128->128->256 -> x2 ----
    int r2 = N2*Kn;
    zero16_kernel<<<cdiv(r2*160,256),256,0,stream>>>(bufA, r2*160);
    gather_sa_kernel<<<cdiv(r2*131,256),256,0,stream>>>(nullptr, x1, 128, pos1, pos2,
                                                        nbr2, N2, Kn, bufA, 160);
    gemm(bufA, 3, r2, bufB, nullptr, 128, 1);
    gemm(bufB, 4, r2, bufA, nullptr, 128, 1);
    gemm(bufA, 5, r2, nullptr, C32, 256, 0);
    ln_kernel<<<cdiv(r2,8),256,0,stream>>>(C32, r2, 256,
        (const float*)d_in[49], (const float*)d_in[48], C32, nullptr);
    maxagg_kernel<<<cdiv(N2*256,256),256,0,stream>>>(C32, d22, N2, Kn, 256, R2, x2);

    // ---- SA3 (global): 259(->288)->512->512->1024, LN, global max -> x3 ----
    zero16_kernel<<<cdiv(N2*288,256),256,0,stream>>>(bufA, N2*288);
    copy16_kernel<<<cdiv(N2*256,256),256,0,stream>>>(x2, 256, bufA, 288, 0, N2, 256);
    copy32to16_kernel<<<cdiv(N2*3,256),256,0,stream>>>(pos2, 3, bufA, 288, 256, N2, 3);
    gemm(bufA, 6, N2, bufB, nullptr, 512, 1);
    gemm(bufB, 7, N2, bufA, nullptr, 512, 1);
    gemm(bufA, 8, N2, nullptr, C32, 1024, 0);
    ln_kernel<<<cdiv(N2,8),256,0,stream>>>(C32, N2, 1024,
        (const float*)d_in[57], (const float*)d_in[56], C32, nullptr);
    gmax_kernel<<<cdiv(1024,256),256,0,stream>>>(C32, N2, 1024, x3);

    // ---- FP1: cat(bcast(x3), x2) [400,1280] -> 256 -> 256 -> 256, LN -> f1 ----
    copy16_kernel<<<cdiv(N2*1024,256),256,0,stream>>>(x3, 0, bufA, 1280, 0, N2, 1024);
    copy16_kernel<<<cdiv(N2*256,256),256,0,stream>>>(x2, 256, bufA, 1280, 1024, N2, 256);
    gemm(bufA,  9, N2, bufB, nullptr, 256, 1);
    gemm(bufB, 10, N2, bufA, nullptr, 256, 1);
    gemm(bufA, 11, N2, nullptr, C32, 256, 0);
    ln_kernel<<<cdiv(N2,8),256,0,stream>>>(C32, N2, 256,
        (const float*)d_in[9], (const float*)d_in[8], nullptr, f1);

    // ---- FP2: cat(interp3(f1), x1) [1600,384] -> 256 -> 256 -> 128, LN -> f2 ----
    knn3_kernel<<<cdiv(N1,256),256,0,stream>>>(pos1, N1, pos2, N2, k3iA, k3wA);
    interp3_kernel<<<cdiv(N1*256,256),256,0,stream>>>(f1, 256, k3iA, k3wA, N1, bufA, 384, 0);
    copy16_kernel<<<cdiv(N1*128,256),256,0,stream>>>(x1, 128, bufA, 384, 256, N1, 128);
    gemm(bufA, 12, N1, bufB, nullptr, 256, 1);
    gemm(bufB, 13, N1, bufA, nullptr, 256, 1);
    gemm(bufA, 14, N1, nullptr, C32, 128, 0);
    ln_kernel<<<cdiv(N1,8),256,0,stream>>>(C32, N1, 128,
        (const float*)d_in[17], (const float*)d_in[16], nullptr, f2);

    // ---- FP3: cat(interp3(f2), x0) [8000,136->160] -> 128 -> 128 -> 128, LN ----
    knn3_kernel<<<cdiv(P,256),256,0,stream>>>(pos0, P, pos1, N1, k3iB, k3wB);
    zero16_kernel<<<cdiv(P*160,256),256,0,stream>>>(bufA, P*160);
    interp3_kernel<<<cdiv(P*128,256),256,0,stream>>>(f2, 128, k3iB, k3wB, P, bufA, 160, 0);
    copy32to16_kernel<<<cdiv(P*8,256),256,0,stream>>>(x0, 8, bufA, 160, 128, P, 8);
    gemm(bufA, 15, P, bufB, nullptr, 128, 1);
    gemm(bufB, 16, P, bufA, nullptr, 128, 1);
    gemm(bufA, 17, P, nullptr, C32, 128, 0);
    ln_kernel<<<cdiv(P,8),256,0,stream>>>(C32, P, 128,
        (const float*)d_in[25], (const float*)d_in[24], nullptr, bufB);  // f3 in bufB

    // ---- head: 128 -> 128 -> 128 -> 128 -> 3 (f32 to d_out) ----
    gemm(bufB, 18, P, bufA, nullptr, 128, 1);
    gemm(bufA, 19, P, bufB, nullptr, 128, 1);
    gemm(bufB, 20, P, bufA, nullptr, 128, 1);
    gemm(bufA, 21, P, nullptr, out + (size_t)b*P*3, 3, 0);
  }
}
